// MLLABlock_46540265619607
// MI455X (gfx1250) — compile-verified, matrix operand order checked by reading, not run on hardware
//
#include <hip/hip_runtime.h>
#include <hip/hip_bf16.h>
#include <stdint.h>

// ---------------- problem constants (fixed by reference) ----------------
constexpr int BB   = 16;
constexpr int HH_  = 48;
constexpr int WW_  = 48;
constexpr int LL   = HH_ * WW_;     // 2304
constexpr int DD   = 384;
constexpr int NHH  = 12;
constexpr int HD   = 32;            // head dim
constexpr int MLPD = 4 * DD;        // 1536
constexpr int NTOK = BB * LL;       // 36864
constexpr int PHALF = DD / 2;       // 192 rope pairs
constexpr int KMAX  = DD / 4;       // 96

typedef __attribute__((ext_vector_type(16))) _Float16 v16h;
typedef __attribute__((ext_vector_type(8)))  float    v8f;

union Frag16 { v16h h; uint4 u[2]; };

// ---------------- generic f32 -> f16 cast ----------------
__global__ __launch_bounds__(256) void cast_f16_kernel(const float* __restrict__ a,
                                                       _Float16* __restrict__ o, size_t n) {
  size_t i = (size_t)blockIdx.x * 256 + threadIdx.x;
  if (i < n) o[i] = (_Float16)a[i];
}

// ---------------- elementwise multiply + cast (att * act_res) ----------------
__global__ __launch_bounds__(256) void mul_cast_kernel(const float* __restrict__ a,
                                                       const float* __restrict__ b,
                                                       _Float16* __restrict__ o, size_t n) {
  size_t i = (size_t)blockIdx.x * 256 + threadIdx.x;
  if (i < n) o[i] = (_Float16)(a[i] * b[i]);
}

// ---------------- depthwise 3x3 conv over HxW token map, channels-last ----------------
// mode 0: out = res + (conv+bias)
// mode 1: out = silu(conv+bias)
// mode 2: out += (conv+bias)
__global__ __launch_bounds__(256) void dwconv_kernel(const float* __restrict__ src,
                                                     const float* __restrict__ wgt,
                                                     const float* __restrict__ bias,
                                                     const float* __restrict__ res,
                                                     float* __restrict__ out, int mode) {
  size_t idx = (size_t)blockIdx.x * 256 + threadIdx.x;
  if (idx >= (size_t)NTOK * DD) return;
  int c = (int)(idx % DD);
  size_t t = idx / DD;
  int l = (int)(t % LL);
  int bimg = (int)(t / LL);
  int hh = l / WW_, ww = l % WW_;
  float acc = bias[c];
  const float* wc = wgt + c * 9;
  #pragma unroll
  for (int dy = -1; dy <= 1; ++dy) {
    int y = hh + dy;
    if ((unsigned)y >= (unsigned)HH_) continue;
    #pragma unroll
    for (int dx = -1; dx <= 1; ++dx) {
      int xw = ww + dx;
      if ((unsigned)xw >= (unsigned)WW_) continue;
      acc += wc[(dy + 1) * 3 + (dx + 1)] *
             src[((size_t)bimg * LL + (size_t)y * WW_ + xw) * DD + c];
    }
  }
  if (mode == 0)      out[idx] = res[idx] + acc;
  else if (mode == 1) out[idx] = acc / (1.f + __expf(-acc));
  else                out[idx] += acc;
}

// ---------------- LayerNorm: one wave32 per token, D=384=12*32 ----------------
__global__ __launch_bounds__(128) void ln_kernel(const float* __restrict__ x,
                                                 const float* __restrict__ w,
                                                 const float* __restrict__ b,
                                                 float* __restrict__ out) {
  int wave = threadIdx.x >> 5, lane = threadIdx.x & 31;
  int tok = blockIdx.x * 4 + wave;
  if (tok >= NTOK) return;
  const float* xr = x + (size_t)tok * DD;
  float vals[12];
  float s = 0.f;
  #pragma unroll
  for (int i = 0; i < 12; ++i) { vals[i] = xr[lane + i * 32]; s += vals[i]; }
  #pragma unroll
  for (int off = 16; off > 0; off >>= 1) s += __shfl_xor(s, off, 32);
  float mean = s * (1.f / DD);
  float sq = 0.f;
  #pragma unroll
  for (int i = 0; i < 12; ++i) { float d = vals[i] - mean; sq += d * d; }
  #pragma unroll
  for (int off = 16; off > 0; off >>= 1) sq += __shfl_xor(sq, off, 32);
  float rstd = rsqrtf(sq * (1.f / DD) + 1e-5f);
  float* orow = out + (size_t)tok * DD;
  #pragma unroll
  for (int i = 0; i < 12; ++i) {
    int c = lane + i * 32;
    orow[c] = (vals[i] - mean) * rstd * w[c] + b[c];
  }
}

// ---------------- RoPE cos/sin tables: (L, D/2) ----------------
__global__ __launch_bounds__(256) void rope_table_kernel(float* __restrict__ cosT,
                                                         float* __restrict__ sinT) {
  int idx = blockIdx.x * 256 + threadIdx.x;
  if (idx >= LL * PHALF) return;
  int p = idx % PHALF;
  int l = idx / PHALF;
  int hh = l / WW_, ww = l % WW_;
  int pp = (p < KMAX) ? p : p - KMAX;
  float pos = (p < KMAX) ? (float)hh : (float)ww;
  float theta = __expf(-((float)pp / (float)KMAX) * 9.210340371976184f); // ln(1e4)
  float ang = pos * theta;
  cosT[idx] = __cosf(ang);
  sinT[idx] = __sinf(ang);
}

// ---------------- WMMA f16 GEMM: out = act(A @ W^T + bias) (+res) ----------------
// A: (M,K) f16 row-major; W: (Nout,K) f16 row-major; block tile 64x128, wave tile 16x64.
// act: 0 none, 1 silu, 2 gelu(exact), 3 elu+1
__global__ __launch_bounds__(256)
void gemm_wmma_kernel(const _Float16* __restrict__ A, const _Float16* __restrict__ Wt,
                      const float* __restrict__ bias, const float* __restrict__ res,
                      float* __restrict__ outF, _Float16* __restrict__ outH,
                      int M, int Nout, int K, int act) {
  int wave = threadIdx.x >> 5;
  int lane = threadIdx.x & 31;
  int wm = wave & 3, wn = wave >> 2;
  int rowBase = blockIdx.x * 64 + wm * 16;
  int colBase = blockIdx.y * 128 + wn * 64;
  int lrow = lane & 15;
  int khalf = (lane >> 4) * 8;           // ISA 16-bit A 16x32 lane layout

  v8f zero = {};
  v8f acc[4];
  #pragma unroll
  for (int j = 0; j < 4; ++j) acc[j] = zero;

  const _Float16* aRow = A + (size_t)(rowBase + lrow) * K + khalf;
  const _Float16* bRow[4];
  #pragma unroll
  for (int j = 0; j < 4; ++j)
    bRow[j] = Wt + (size_t)(colBase + j * 16 + lrow) * K + khalf;

  for (int k = 0; k < K; k += 32) {
    Frag16 fa;
    fa.u[0] = *(const uint4*)(aRow + k);
    fa.u[1] = *(const uint4*)(aRow + k + 16);
    #pragma unroll
    for (int j = 0; j < 4; ++j) {
      Frag16 fb;
      fb.u[0] = *(const uint4*)(bRow[j] + k);
      fb.u[1] = *(const uint4*)(bRow[j] + k + 16);
      acc[j] = __builtin_amdgcn_wmma_f32_16x16x32_f16(
          false, fa.h, false, fb.h, (short)0, acc[j], false, false);
    }
  }

  int orow = rowBase + (lane >> 4) * 8;  // 32-bit C/D layout: vgpr r -> M=r / r+8
  #pragma unroll
  for (int j = 0; j < 4; ++j) {
    int col = colBase + j * 16 + lrow;
    float bc = bias ? bias[col] : 0.f;
    #pragma unroll
    for (int r = 0; r < 8; ++r) {
      float v = acc[j][r] + bc;
      if (act == 1)      v = v / (1.f + __expf(-v));                       // silu
      else if (act == 2) v = 0.5f * v * (1.f + erff(v * 0.7071067811865476f)); // gelu
      else if (act == 3) v = (v > 0.f) ? (v + 1.f) : __expf(v);            // elu+1
      size_t off = (size_t)(orow + r) * Nout + col;
      if (res) v += res[off];
      if (outF) outF[off] = v;
      else      outH[off] = (_Float16)v;
    }
  }
}

// ---------------- kmean: per (b,h) mean of k over L ----------------
__global__ __launch_bounds__(32) void kmean_kernel(const float* __restrict__ qk,
                                                   float* __restrict__ kmean) {
  int bh = blockIdx.x;
  int d = threadIdx.x;
  int b = bh / NHH, h = bh % NHH;
  const float* base = qk + (size_t)b * LL * (2 * DD) + DD + h * HD + d;
  float s = 0.f;
  for (int n = 0; n < LL; ++n) s += base[(size_t)n * (2 * DD)];
  kmean[bh * HD + d] = s * (1.f / LL);
}

// ---------------- kv = (1/L) * k_rope^T @ v, per (b,h), LDS-staged ----------------
constexpr int KV_CH = 8;
__global__ __launch_bounds__(256)
void kv_kernel(const float* __restrict__ qk, const float* __restrict__ xi,
               const float* __restrict__ cosT, const float* __restrict__ sinT,
               float* __restrict__ kv) {
  __shared__ float lk[KV_CH][HD];
  __shared__ float lv[KV_CH][HD];
  int bh = blockIdx.x;
  int b = bh / NHH, h = bh % NHH;
  int t = threadIdx.x;
  int d = t >> 3;
  int e4 = (t & 7) * 4;
  float a0 = 0.f, a1 = 0.f, a2 = 0.f, a3 = 0.f;
  for (int n0 = 0; n0 < LL; n0 += KV_CH) {
    if (t < 128) {                       // rope'd k loaders: 8 rows x 16 pairs
      int r = t >> 4, pr = t & 15;
      int n = n0 + r;
      size_t row = (size_t)b * LL + n;
      float ke = qk[row * (2 * DD) + DD + h * HD + 2 * pr];
      float ko = qk[row * (2 * DD) + DD + h * HD + 2 * pr + 1];
      int p = h * (HD / 2) + pr;
      float cs = cosT[n * PHALF + p], sn = sinT[n * PHALF + p];
      lk[r][2 * pr]     = cs * ke - sn * ko;
      lk[r][2 * pr + 1] = sn * ke + cs * ko;
    } else {                             // v loaders: 8 rows x 32 vals / 128 threads
      int u = t - 128;
      int r = u >> 4, ee = (u & 15) * 2;
      int n = n0 + r;
      size_t row = (size_t)b * LL + n;
      lv[r][ee]     = xi[row * DD + h * HD + ee];
      lv[r][ee + 1] = xi[row * DD + h * HD + ee + 1];
    }
    __syncthreads();
    #pragma unroll
    for (int r = 0; r < KV_CH; ++r) {
      float kr = lk[r][d];
      a0 += kr * lv[r][e4 + 0];
      a1 += kr * lv[r][e4 + 1];
      a2 += kr * lv[r][e4 + 2];
      a3 += kr * lv[r][e4 + 3];
    }
    __syncthreads();
  }
  float inv = 1.f / LL;
  float* o = kv + (size_t)bh * (HD * HD) + d * HD + e4;
  o[0] = a0 * inv; o[1] = a1 * inv; o[2] = a2 * inv; o[3] = a3 * inv;
}

// ---------------- att = z * (q_rope @ kv), one wave32 per (token, head) ----------------
__global__ __launch_bounds__(256)
void att_kernel(const float* __restrict__ qk, const float* __restrict__ kmean,
                const float* __restrict__ kv, const float* __restrict__ cosT,
                const float* __restrict__ sinT, float* __restrict__ att) {
  int wid = (int)(((size_t)blockIdx.x * 256 + threadIdx.x) >> 5);
  int lane = threadIdx.x & 31;
  if (wid >= NTOK * NHH) return;
  int h = wid % NHH;
  int tok = wid / NHH;
  int b = tok / LL, l = tok % LL;
  int c = h * HD + lane;
  float qd = qk[(size_t)tok * (2 * DD) + c];           // elu+1 already applied
  float km = kmean[(b * NHH + h) * HD + lane];
  float zs = qd * km;
  #pragma unroll
  for (int off = 16; off > 0; off >>= 1) zs += __shfl_xor(zs, off, 32);
  float z = 1.f / (zs + 1e-6f);
  float qp = __shfl_xor(qd, 1, 32);
  int p = c >> 1;
  float cs = cosT[l * PHALF + p], sn = sinT[l * PHALF + p];
  float qr = (lane & 1) ? (sn * qp + cs * qd) : (cs * qd - sn * qp);
  const float* kvb = kv + (size_t)(b * NHH + h) * (HD * HD);
  float accv = 0.f;
  #pragma unroll 8
  for (int dd = 0; dd < HD; ++dd) {
    float q_d = __shfl(qr, dd, 32);
    accv += q_d * kvb[dd * HD + lane];
  }
  att[(size_t)tok * DD + c] = accv * z;
}

// ================= host orchestration =================
extern "C" void kernel_launch(void* const* d_in, const int* in_sizes, int n_in,
                              void* d_out, int out_size, void* d_ws, size_t ws_size,
                              hipStream_t stream) {
  const float* x        = (const float*)d_in[0];
  const float* cpe1_w   = (const float*)d_in[3];
  const float* cpe1_b   = (const float*)d_in[4];
  const float* norm1_w  = (const float*)d_in[5];
  const float* norm1_b  = (const float*)d_in[6];
  const float* in_w     = (const float*)d_in[7];
  const float* in_b     = (const float*)d_in[8];
  const float* act_w    = (const float*)d_in[9];
  const float* act_b    = (const float*)d_in[10];
  const float* dwc_w    = (const float*)d_in[11];
  const float* dwc_b    = (const float*)d_in[12];
  const float* qk_w     = (const float*)d_in[13];
  const float* qk_b     = (const float*)d_in[14];
  const float* lepe_w   = (const float*)d_in[15];
  const float* lepe_b   = (const float*)d_in[16];
  const float* out_w    = (const float*)d_in[17];
  const float* out_b    = (const float*)d_in[18];
  const float* cpe2_w   = (const float*)d_in[19];
  const float* cpe2_b   = (const float*)d_in[20];
  const float* norm2_w  = (const float*)d_in[21];
  const float* norm2_b  = (const float*)d_in[22];
  const float* fc1_w    = (const float*)d_in[23];
  const float* fc1_b    = (const float*)d_in[24];
  const float* fc2_w    = (const float*)d_in[25];
  const float* fc2_b    = (const float*)d_in[26];

  uint8_t* ws = (uint8_t*)d_ws;
  const size_t ND = (size_t)NTOK * DD;                 // elems
  const size_t SZ = ND * sizeof(float);                // 56,623,104 bytes (256-aligned)
  size_t off = 0;
  float* B0 = (float*)(ws + off); off += SZ;           // X1 / X3
  float* B1 = (float*)(ws + off); off += SZ;           // XN / ATT / XN2
  float* B2 = (float*)(ws + off); off += SZ;           // ACT / X2
  float* B3 = (float*)(ws + off); off += SZ;           // XI0
  float* B4 = (float*)(ws + off); off += SZ;           // XI
  float* B5 = (float*)(ws + off); off += 2 * SZ;       // QK (N x 2D)
  _Float16* F0 = (_Float16*)(ws + off); off += SZ / 2; // f16 staging (N x D)
  _Float16* F1 = (_Float16*)(ws + off); off += 2 * SZ; // f16 H1 (N x MLP)
  _Float16* Win  = (_Float16*)(ws + off); off += (size_t)DD * DD * 2;
  _Float16* Wact = (_Float16*)(ws + off); off += (size_t)DD * DD * 2;
  _Float16* Wqk  = (_Float16*)(ws + off); off += (size_t)2 * DD * DD * 2;
  _Float16* Wout = (_Float16*)(ws + off); off += (size_t)DD * DD * 2;
  _Float16* Wfc1 = (_Float16*)(ws + off); off += (size_t)MLPD * DD * 2;
  _Float16* Wfc2 = (_Float16*)(ws + off); off += (size_t)DD * MLPD * 2;
  float* ROPC  = (float*)(ws + off); off += (size_t)LL * PHALF * 4;
  float* ROPS  = (float*)(ws + off); off += (size_t)LL * PHALF * 4;
  float* KMEAN = (float*)(ws + off); off += (size_t)BB * NHH * HD * 4;
  float* KV    = (float*)(ws + off); off += (size_t)BB * NHH * HD * HD * 4;
  (void)ws_size; (void)in_sizes; (void)n_in; (void)out_size;

  auto cast = [&](const float* s, _Float16* d, size_t n) {
    cast_f16_kernel<<<dim3((unsigned)((n + 255) / 256)), dim3(256), 0, stream>>>(s, d, n);
  };
  auto gemm = [&](const _Float16* A, const _Float16* Wt, const float* bias,
                  const float* res, float* oF, _Float16* oH, int Nout, int K, int act) {
    gemm_wmma_kernel<<<dim3(NTOK / 64, Nout / 128), dim3(256), 0, stream>>>(
        A, Wt, bias, res, oF, oH, NTOK, Nout, K, act);
  };
  const unsigned EL_BLK = (unsigned)((ND + 255) / 256);          // 55296

  // weights -> f16, rope tables
  cast(in_w,  Win,  (size_t)DD * DD);
  cast(act_w, Wact, (size_t)DD * DD);
  cast(qk_w,  Wqk,  (size_t)2 * DD * DD);
  cast(out_w, Wout, (size_t)DD * DD);
  cast(fc1_w, Wfc1, (size_t)MLPD * DD);
  cast(fc2_w, Wfc2, (size_t)DD * MLPD);
  rope_table_kernel<<<dim3((LL * PHALF + 255) / 256), dim3(256), 0, stream>>>(ROPC, ROPS);

  // 1) x = x + cpe1(x)  -> B0 (shortcut)
  dwconv_kernel<<<EL_BLK, 256, 0, stream>>>(x, cpe1_w, cpe1_b, x, B0, 0);
  // 2) LN1 -> B1, cast -> F0
  ln_kernel<<<NTOK / 4, 128, 0, stream>>>(B0, norm1_w, norm1_b, B1);
  cast(B1, F0, ND);
  // 3) act_res = silu(XN @ act^T + b) -> B2
  gemm(F0, Wact, act_b, nullptr, B2, nullptr, DD, DD, 1);
  // 4) xi0 = XN @ in^T + b -> B3
  gemm(F0, Win, in_b, nullptr, B3, nullptr, DD, DD, 0);
  // 5) xi = silu(dwc(xi0)) -> B4, cast -> F0
  dwconv_kernel<<<EL_BLK, 256, 0, stream>>>(B3, dwc_w, dwc_b, nullptr, B4, 1);
  cast(B4, F0, ND);
  // 6) qk = elu(xi @ qk^T + b) + 1 -> B5 (N x 2D)
  gemm(F0, Wqk, qk_b, nullptr, B5, nullptr, 2 * DD, DD, 3);
  // 7) attention state
  kmean_kernel<<<BB * NHH, 32, 0, stream>>>(B5, KMEAN);
  kv_kernel<<<BB * NHH, 256, 0, stream>>>(B5, B4, ROPC, ROPS, KV);
  att_kernel<<<dim3((unsigned)(((size_t)NTOK * NHH * 32 + 255) / 256)), dim3(256), 0,
               stream>>>(B5, KMEAN, KV, ROPC, ROPS, B1);
  // 8) att += lepe(xi)
  dwconv_kernel<<<EL_BLK, 256, 0, stream>>>(B4, lepe_w, lepe_b, nullptr, B1, 2);
  // 9) F0 = f16(att * act_res)
  mul_cast_kernel<<<EL_BLK, 256, 0, stream>>>(B1, B2, F0, ND);
  // 10) x2 = shortcut + (F0 @ out^T + b) -> B2
  gemm(F0, Wout, out_b, B0, B2, nullptr, DD, DD, 0);
  // 11) x3 = x2 + cpe2(x2) -> B0
  dwconv_kernel<<<EL_BLK, 256, 0, stream>>>(B2, cpe2_w, cpe2_b, B2, B0, 0);
  // 12) LN2 -> B1, cast -> F0
  ln_kernel<<<NTOK / 4, 128, 0, stream>>>(B0, norm2_w, norm2_b, B1);
  cast(B1, F0, ND);
  // 13) h1 = gelu(XN2 @ fc1^T + b) -> F1 (f16)
  gemm(F0, Wfc1, fc1_b, nullptr, nullptr, F1, MLPD, DD, 2);
  // 14) out = x3 + (h1 @ fc2^T + b) -> d_out
  gemm(F1, Wfc2, fc2_b, B0, (float*)d_out, nullptr, DD, MLPD, 0);
}
